// PermutedResidualAttentionBlock_74612171866118
// MI455X (gfx1250) — matrix operand order repaired and checked by reading, not compiled, not run on hardware
//
#include <hip/hip_runtime.h>

// ---- MI455X / gfx1250 (CDNA5, wave32, WMMA) implementation ----

typedef __attribute__((ext_vector_type(16))) __bf16        v16bf;
typedef __attribute__((ext_vector_type(8)))  float         v8f;
typedef __attribute__((ext_vector_type(4)))  float         f32x4;
typedef __attribute__((ext_vector_type(4)))  unsigned int  u32x4;

#define DIMD   256
#define NSEQ   8192
#define BATCH  8
#define WIN    512
#define NWIN   (NSEQ / WIN)          // 16
#define ROWS   (BATCH * NSEQ)        // 65536

__device__ __forceinline__ unsigned short f2bf(float f) {
  unsigned u = __builtin_bit_cast(unsigned, f);
  u += 0x7FFFu + ((u >> 16) & 1u);   // round-to-nearest-even
  return (unsigned short)(u >> 16);
}

// 16x32 bf16 WMMA operand fragment (8 VGPRs / lane).
union BFfrag {
  v16bf          v;
  f32x4          f4[2];   // two 16-byte chunks (8 bf16 each)
  unsigned       u[8];
  unsigned short h[16];
};

// ---------------- kernel 1: weights f32 -> bf16 ----------------
__global__ void k_w2bf(const float* __restrict__ Wq, const float* __restrict__ Wk,
                       const float* __restrict__ Wv, unsigned short* __restrict__ Wbf) {
  int i = blockIdx.x * blockDim.x + threadIdx.x;   // 0..65535
  Wbf[i]          = f2bf(Wq[i]);
  Wbf[i + 65536]  = f2bf(Wk[i]);
  Wbf[i + 131072] = f2bf(Wv[i]);
}

// ---------------- kernel 2: x (B,D,N) f32 -> xbf (B,N,D) bf16 ----------------
__global__ void k_transpose(const float* __restrict__ x, unsigned short* __restrict__ xbf) {
  __shared__ float tile[32][33];
  const int b = blockIdx.z;
  const int n0 = blockIdx.x * 32, d0 = blockIdx.y * 32;
  const int tx = threadIdx.x & 31, ty = threadIdx.x >> 5;  // ty 0..7
  const float* src = x + (size_t)b * DIMD * NSEQ;
#pragma unroll
  for (int i = 0; i < 4; ++i)
    tile[ty + 8 * i][tx] = src[(size_t)(d0 + ty + 8 * i) * NSEQ + n0 + tx];
  __syncthreads();
  unsigned short* dst = xbf + (size_t)b * NSEQ * DIMD;
#pragma unroll
  for (int i = 0; i < 4; ++i) {
    const int r = ty + 8 * i;
    dst[(size_t)(n0 + r) * DIMD + d0 + tx] = f2bf(tile[tx][r]);
  }
}

// ---------------- kernel 3: fused QKV projection GEMM (bf16 WMMA) ----------
// grid.x = ROWS/32 row-blocks. 256 threads = 8 waves, wave grid 2(M)x4(N):
// workgroup computes 32 rows x 256 cols for ALL THREE of q,k,v — each A
// fragment feeds 12 WMMAs. __launch_bounds__(256,1) relaxes the register
// pressure target so the 4-slot rotating B buffer survives RA, keeping 3
// weight loads in flight ahead of every WMMA.
// q,k stored (B,N,D) bf16 row-major; v stored transposed (B,D,N) bf16.
__global__ void __launch_bounds__(256, 1)
k_qkv(const unsigned short* __restrict__ xbf,
      const unsigned short* __restrict__ Wbf,
      const float* __restrict__ bq, const float* __restrict__ bk,
      const float* __restrict__ bv,
      unsigned short* __restrict__ qbf,
      unsigned short* __restrict__ kbf,
      unsigned short* __restrict__ vT) {
  const int rb = blockIdx.x;
  const int tid = threadIdx.x, lane = tid & 31, wv = tid >> 5;
  const int wm = wv >> 2, wn = wv & 3;
  const int l15 = lane & 15, hi = lane >> 4;      // 0 or 1
  const int c0 = hi ? 16 : 0;                     // A-fragment K sub-chunk
  const int rowA = rb * 32 + wm * 16 + l15;

  v8f acc[12] = {};                                // [which*4 + nt]
  for (int k0 = 0; k0 < DIMD; k0 += 32) {
    BFfrag a;
    const unsigned short* ap = xbf + (size_t)rowA * DIMD + k0 + c0;
    a.f4[0] = *(const f32x4*)ap;                  // K = k0+c0 .. +7
    a.f4[1] = *(const f32x4*)(ap + 8);            // K = k0+c0+8 .. +15

    const unsigned short* Wbase =
        Wbf + (size_t)(wn * 64 + l15) * DIMD + k0 + (hi ? 16 : 0);
    // B fragment pointer for flat index idx = which*4 + nt
    auto bptr = [&](int idx) {
      const int which = idx >> 2, nt = idx & 3;
      return Wbase + (size_t)which * 65536 + (size_t)nt * 16 * DIMD;
    };
    // 4-slot rotating buffer: 3 loads in flight ahead of each WMMA.
    BFfrag bb[4];
    bb[0].v = *(const v16bf*)bptr(0);
    bb[1].v = *(const v16bf*)bptr(1);
    bb[2].v = *(const v16bf*)bptr(2);
#pragma unroll
    for (int idx = 0; idx < 12; ++idx) {
      if (idx < 9) bb[(idx + 3) & 3].v = *(const v16bf*)bptr(idx + 3);
      acc[idx] = __builtin_amdgcn_wmma_f32_16x16x32_bf16(
          false, a.v, false, bb[idx & 3].v, (short)0, acc[idx], false, false);
    }
  }

  // q and k: (B,N,D) row-major bf16 stores
#pragma unroll
  for (int which = 0; which < 2; ++which) {
    unsigned short* dst = (which == 0) ? qbf : kbf;
    const float* bias = (which == 0) ? bq : bk;
#pragma unroll
    for (int nt = 0; nt < 4; ++nt) {
      const int ncol = wn * 64 + nt * 16 + l15;
      const float bb = bias[ncol];
#pragma unroll
      for (int vg = 0; vg < 8; ++vg) {
        const int row = rb * 32 + wm * 16 + vg + 8 * hi;  // C layout: M = vg + 8*hi
        dst[(size_t)row * DIMD + ncol] = f2bf(acc[which * 4 + nt][vg] + bb);
      }
    }
  }
  // v transposed: vT[b][d][n]; lane owns 8 consecutive n -> one b128 store/tile
  {
    const int rowbase = rb * 32 + wm * 16 + 8 * hi;
    const int b = rowbase >> 13, nrel = rowbase & (NSEQ - 1);
#pragma unroll
    for (int nt = 0; nt < 4; ++nt) {
      const int d = wn * 64 + nt * 16 + l15;
      const float bb = bv[d];
      u32x4 p;
#pragma unroll
      for (int j = 0; j < 4; ++j) {
        const unsigned lo = f2bf(acc[8 + nt][2 * j] + bb);
        const unsigned hh = f2bf(acc[8 + nt][2 * j + 1] + bb);
        p[j] = lo | (hh << 16);
      }
      *(u32x4*)(vT + ((size_t)b * DIMD + d) * NSEQ + nrel) = p;
    }
  }
}

// ---------------- kernel 4: local attention + residual + output transpose ----
// One workgroup per (b, window w, 64-query chunk qc). 256 KB dynamic LDS holds
// the 64x1024 f32 score/probability tile (fits 320 KB WGP LDS); that already
// limits us to 1 workgroup/WGP, so claim the full register budget too.
__global__ void __launch_bounds__(256, 1)
k_attn(const unsigned short* __restrict__ qbf,
       const unsigned short* __restrict__ kbf,
       const unsigned short* __restrict__ vT,
       const float* __restrict__ x,
       float* __restrict__ out) {
  extern __shared__ float S[];                       // [64][1024]
  const int bid = blockIdx.x;
  const int b = bid >> 7;                            // / (NWIN*8)
  const int w = (bid >> 3) & (NWIN - 1);
  const int qc = bid & 7;
  const int tid = threadIdx.x, lane = tid & 31, wv = tid >> 5;
  const int l15 = lane & 15, hi = lane >> 4;
  const int c0 = hi ? 16 : 0;
  const int qrow0 = w * WIN + qc * 64;               // within-batch query base
  const int kbase = w * WIN - WIN;                   // first key pos (may be <0)

  // ---- phase 1: S = (Q @ K^T) * scale ----
  {
    const int m = wv >> 1;                           // 0..3  (16 queries)
    const int nb = (wv & 1) * 32;                    // 32 key-tiles of 16
    const int qrow = qrow0 + m * 16 + l15;
    BFfrag qf[8];
#pragma unroll
    for (int ks = 0; ks < 8; ++ks) {
      const unsigned short* ap = qbf + ((size_t)b * NSEQ + qrow) * DIMD + ks * 32 + c0;
      qf[ks].f4[0] = *(const f32x4*)ap;
      qf[ks].f4[1] = *(const f32x4*)(ap + 8);
    }
    for (int nn = nb; nn < nb + 32; ++nn) {
      int kseq = kbase + nn * 16 + l15;
      if (kseq < 0) kseq = 0;                        // clamped; masked to prob 0 later
      const unsigned short* kp = kbf + ((size_t)b * NSEQ + kseq) * DIMD + (hi ? 16 : 0);
      __builtin_prefetch((const void*)(kp + 16 * DIMD), 0, 1);  // next key-tile row
      // Preload all 8 K fragments (one load clause), then back-to-back WMMAs.
      BFfrag kf[8];
#pragma unroll
      for (int ks = 0; ks < 8; ++ks) kf[ks].v = *(const v16bf*)(kp + ks * 32);
      v8f acc = {};
#pragma unroll
      for (int ks = 0; ks < 8; ++ks)
        acc = __builtin_amdgcn_wmma_f32_16x16x32_bf16(false, qf[ks].v, false, kf[ks].v,
                                                      (short)0, acc, false, false);
#pragma unroll
      for (int vg = 0; vg < 8; ++vg) {
        const int Mr = m * 16 + vg + 8 * hi;
        S[Mr * 1024 + nn * 16 + l15] = acc[vg] * 0.0625f;   // 256^-0.5
      }
    }
  }
  __syncthreads();

  // ---- masked softmax over the 64 rows (4 threads per row, f32x4 LDS ops) ----
  {
    const int row = tid >> 2;
    const int part = tid & 3;
    const int cb = part * 256;
    const int p = qrow0 + row;                       // query position
    float* sr = S + row * 1024 + cb;
    float mx = -3.0e38f;
    for (int c = 0; c < 256; c += 4) {
      const f32x4 sv = *(const f32x4*)(sr + c);
#pragma unroll
      for (int j = 0; j < 4; ++j) {
        const int kpos = kbase + cb + c + j;
        const bool ok = (kpos >= 0) && (p >= kpos);
        mx = fmaxf(mx, ok ? sv[j] : -3.0e38f);
      }
    }
    mx = fmaxf(mx, __shfl_xor(mx, 1, 4));
    mx = fmaxf(mx, __shfl_xor(mx, 2, 4));
    float sum = 0.f;
    for (int c = 0; c < 256; c += 4) {
      const f32x4 sv = *(const f32x4*)(sr + c);
      f32x4 ev;
#pragma unroll
      for (int j = 0; j < 4; ++j) {
        const int kpos = kbase + cb + c + j;
        const bool ok = (kpos >= 0) && (p >= kpos);
        ev[j] = ok ? __expf(sv[j] - mx) : 0.f;
        sum += ev[j];
      }
      *(f32x4*)(sr + c) = ev;
    }
    sum += __shfl_xor(sum, 1, 4);
    sum += __shfl_xor(sum, 2, 4);
    const float inv = 1.f / sum;
    for (int c = 0; c < 256; c += 4) {
      f32x4 sv = *(const f32x4*)(sr + c);
#pragma unroll
      for (int j = 0; j < 4; ++j) sv[j] *= inv;
      *(f32x4*)(sr + c) = sv;
    }
  }
  __syncthreads();

  // ---- phase 2: O = P @ V, residual add, transposed f32 store ----
  {
    const int m = wv >> 1;                           // query tile
    const int nb = (wv & 1) * 8;                     // 8 output-dim tiles
    const int rowl = m * 16 + l15;
    v8f oacc[8] = {};
    for (int k0 = 0; k0 < 1024; k0 += 32) {
      BFfrag a;                                      // P fragment: f32 LDS -> bf16
      const float* sp = S + rowl * 1024 + k0 + c0;
      const f32x4 a0 = *(const f32x4*)sp;
      const f32x4 a1 = *(const f32x4*)(sp + 4);
      const f32x4 a2 = *(const f32x4*)(sp + 8);
      const f32x4 a3 = *(const f32x4*)(sp + 12);
      a.u[0] = (unsigned)f2bf(a0[0]) | ((unsigned)f2bf(a0[1]) << 16);
      a.u[1] = (unsigned)f2bf(a0[2]) | ((unsigned)f2bf(a0[3]) << 16);
      a.u[2] = (unsigned)f2bf(a1[0]) | ((unsigned)f2bf(a1[1]) << 16);
      a.u[3] = (unsigned)f2bf(a1[2]) | ((unsigned)f2bf(a1[3]) << 16);
      a.u[4] = (unsigned)f2bf(a2[0]) | ((unsigned)f2bf(a2[1]) << 16);
      a.u[5] = (unsigned)f2bf(a2[2]) | ((unsigned)f2bf(a2[3]) << 16);
      a.u[6] = (unsigned)f2bf(a3[0]) | ((unsigned)f2bf(a3[1]) << 16);
      a.u[7] = (unsigned)f2bf(a3[2]) | ((unsigned)f2bf(a3[3]) << 16);
      int kg = kbase + k0 + (hi ? 16 : 0);
      if (kg < 0) kg = 0;                            // clamped; P rows are 0 there
      const unsigned short* vrow = vT + (size_t)b * DIMD * NSEQ + kg;
      __builtin_prefetch((const void*)(vrow + (size_t)(nb * 16 + l15) * NSEQ +
                                       ((kg + 32 < NSEQ) ? 32 : 0)), 0, 1);
      // 4-slot rotating V buffer: 3 loads in flight ahead of each WMMA.
      BFfrag vf[4];
      vf[0].v = *(const v16bf*)(vrow + (size_t)((nb + 0) * 16 + l15) * NSEQ);
      vf[1].v = *(const v16bf*)(vrow + (size_t)((nb + 1) * 16 + l15) * NSEQ);
      vf[2].v = *(const v16bf*)(vrow + (size_t)((nb + 2) * 16 + l15) * NSEQ);
#pragma unroll
      for (int i = 0; i < 8; ++i) {
        if (i < 5)
          vf[(i + 3) & 3].v =
              *(const v16bf*)(vrow + (size_t)((nb + i + 3) * 16 + l15) * NSEQ);
        oacc[i] = __builtin_amdgcn_wmma_f32_16x16x32_bf16(
            false, a.v, false, vf[i & 3].v, (short)0, oacc[i], false, false);
      }
    }
    const int np0 = qrow0 + m * 16 + 8 * hi;         // 8 consecutive n per lane
#pragma unroll
    for (int i = 0; i < 8; ++i) {
      const int d = (nb + i) * 16 + l15;
      const size_t base = ((size_t)b * DIMD + d) * NSEQ + np0;
      const f32x4 r0 = *(const f32x4*)(x + base);
      const f32x4 r1 = *(const f32x4*)(x + base + 4);
      f32x4 o0, o1;
#pragma unroll
      for (int j = 0; j < 4; ++j) {
        o0[j] = r0[j] + oacc[i][j];
        o1[j] = r1[j] + oacc[i][4 + j];
      }
      *(f32x4*)(out + base) = o0;
      *(f32x4*)(out + base + 4) = o1;
    }
  }
}

// ---------------- launch ----------------
extern "C" void kernel_launch(void* const* d_in, const int* in_sizes, int n_in,
                              void* d_out, int out_size, void* d_ws, size_t ws_size,
                              hipStream_t stream) {
  (void)in_sizes; (void)n_in; (void)out_size; (void)ws_size;
  const float* x  = (const float*)d_in[0];
  const float* Wq = (const float*)d_in[1];
  const float* bq = (const float*)d_in[2];
  const float* Wk = (const float*)d_in[3];
  const float* bk = (const float*)d_in[4];
  const float* Wv = (const float*)d_in[5];
  const float* bv = (const float*)d_in[6];
  float* out = (float*)d_out;

  char* ws = (char*)d_ws;
  unsigned short* Wbf = (unsigned short*)ws;                          // 3*64K bf16
  unsigned short* xbf = (unsigned short*)(ws + 393216);               // (B,N,D)
  unsigned short* qbf = xbf + (size_t)ROWS * DIMD;                    // (B,N,D)
  unsigned short* kbf = qbf + (size_t)ROWS * DIMD;                    // (B,N,D)
  unsigned short* vT  = kbf + (size_t)ROWS * DIMD;                    // (B,D,N)

  k_w2bf<<<dim3(256), dim3(256), 0, stream>>>(Wq, Wk, Wv, Wbf);
  k_transpose<<<dim3(NSEQ / 32, DIMD / 32, BATCH), dim3(256), 0, stream>>>(x, xbf);
  k_qkv<<<dim3(ROWS / 32), dim3(256), 0, stream>>>(xbf, Wbf, bq, bk, bv, qbf, kbf, vT);
  k_attn<<<dim3(BATCH * NWIN * 8), dim3(256), 64 * 1024 * 4, stream>>>(qbf, kbf, vT, x, out);
}
